// TtMambaSSM_39582418600449
// MI455X (gfx1250) — matrix operand order repaired and checked by reading, not compiled, not run on hardware
//
#include <hip/hip_runtime.h>

// Problem dimensions (fixed by the reference)
#define U_DIM 256
#define D_DIM 5120
#define N_DIM 16
#define R_DIM 160

typedef __attribute__((ext_vector_type(16))) __bf16 v16bf;
typedef __attribute__((ext_vector_type(8)))  float  v8f;
typedef __attribute__((ext_vector_type(4)))  float  f4;

// ---- split-precision bf16 packing ---------------------------------------
// f32 = hi + lo with hi = bf16(f), lo = bf16(f - hi). A*B over {hi,lo}
// keeping the three largest terms reproduces f32 GEMM accuracy on the
// bf16 WMMA pipe (the lo*lo term is < 2^-16 relative).

struct FragPair { v16bf hi, lo; };

__device__ __forceinline__ FragPair pack_split(const float* __restrict__ f) {
    union Cvt { v16bf v; __bf16 e[16]; };
    Cvt hi, lo;
#pragma unroll
    for (int j = 0; j < 16; ++j) {
        const float  x = f[j];
        const __bf16 h = (__bf16)x;        // native v_cvt, RNE
        hi.e[j] = h;
        lo.e[j] = (__bf16)(x - (float)h);  // residual
    }
    return {hi.v, lo.v};
}

__device__ __forceinline__ v8f wmma3(const FragPair& a, const FragPair& b, v8f acc) {
    acc = __builtin_amdgcn_wmma_f32_16x16x32_bf16(
        false, a.hi, false, b.hi, (short)0, acc, false, false);
    acc = __builtin_amdgcn_wmma_f32_16x16x32_bf16(
        false, a.lo, false, b.hi, (short)0, acc, false, false);
    acc = __builtin_amdgcn_wmma_f32_16x16x32_bf16(
        false, a.hi, false, b.lo, (short)0, acc, false, false);
    return acc;
}

// A-fragment: 16x32 tile of a row-major f32 matrix (ld = row stride), start
// (r0,k0). ISA layout (16-bit A 16x32): lane l (m=l&15, half=l>>4) holds
// K = half*8 + {0..7} in vector elems 0..7 and K = 16 + half*8 + {0..7} in
// elems 8..15 (two bf16 per VGPR).
__device__ __forceinline__ FragPair load_a_frag(const float* __restrict__ A,
                                                int ld, int r0, int k0, int lane) {
    const int m = lane & 15, half = lane >> 4;
    const float* p = A + (size_t)(r0 + m) * ld + k0 + half * 8;
    const f4 a0 = *reinterpret_cast<const f4*>(p);
    const f4 a1 = *reinterpret_cast<const f4*>(p + 4);
    const f4 a2 = *reinterpret_cast<const f4*>(p + 16);
    const f4 a3 = *reinterpret_cast<const f4*>(p + 20);
    float t[16];
#pragma unroll
    for (int j = 0; j < 4; ++j) {
        t[j] = a0[j]; t[4 + j] = a1[j]; t[8 + j] = a2[j]; t[12 + j] = a3[j];
    }
    return pack_split(t);
}

// B-fragment: 32x16 tile of a row-major f32 matrix (ld = row stride), tile
// start (k0 row, c0 col). ISA layout (16-bit B 32x16): lane l (n=l&15,
// half=l>>4) holds K = half*8 + {0..7}, 16 + half*8 + {0..7} for column c0+n. With c0 64B-
// aligned, each dword load covers two fully-used 64B segments.
__device__ __forceinline__ FragPair load_b_frag(const float* __restrict__ W,
                                                int ld, int k0, int c0, int lane) {
    const int n = lane & 15, half = lane >> 4;
    const float* p = W + (size_t)(k0 + half * 8) * ld + c0 + n;
    float t[16];
#pragma unroll
    for (int j = 0; j < 16; ++j)
        t[j] = p[(size_t)(j < 8 ? j : j + 8) * ld];
    return pack_split(t);
}

// ---- kernel 0: zero the split-K accumulators -----------------------------

__global__ void k_zero(float* __restrict__ p, int n) {
    int i = blockIdx.x * blockDim.x + threadIdx.x;
    if (i < n) p[i] = 0.0f;
}

// ---- kernel 1: T = x@W1 ([U,R]) and Bm = x@Wb ([U,N]), split-K + atomics --
// tasks = 16 u-tiles * 11 col-tiles (10 for T, 1 for Bm) * 8 k-splits = 1408
// grid 176 blocks * 8 waves; all task selection is wave-uniform so EXEC
// stays all-ones for WMMA.

__global__ void k_gemm1(const float* __restrict__ x,    // [U,D]
                        const float* __restrict__ W1,   // [D,R]
                        const float* __restrict__ Wb,   // [D,N]
                        float* __restrict__ T,          // [U,R] (ws)
                        float* __restrict__ Bm) {       // [U,N] (ws)
    const int lane = threadIdx.x & 31;
    const int wid  = threadIdx.x >> 5;
    const int task = blockIdx.x * 8 + wid;      // wave-uniform
    const int u  = task & 15;                   // 16 u-tiles
    const int t2 = task >> 4;                   // 0..87
    const int c  = t2 % 11;                     // 11 col-tiles
    const int ks = t2 / 11;                     // 8 k-splits
    const int u0 = u * 16;
    const int c0 = c * 16;
    const int k_begin = ks * (D_DIM / 8);       // 640-deep K slice
    const int k_end   = k_begin + (D_DIM / 8);

    const bool  isT = (c0 < R_DIM);             // wave-uniform branch
    const float* W  = isT ? W1 : Wb;
    const int   ldw = isT ? R_DIM : N_DIM;
    const int   cc0 = isT ? c0 : (c0 - R_DIM);

    v8f acc = {};
    for (int k = k_begin; k < k_end; k += 32) {
        FragPair a = load_a_frag(x, D_DIM, u0, k, lane);
        FragPair b = load_b_frag(W, ldw, k, cc0, lane);
        acc = wmma3(a, b, acc);
    }

    // C layout: VGPR r, lane l -> row m = r + 8*(l>>4), col n = l&15
    float* outp   = isT ? T : Bm;
    const int ldo = ldw;
    const int n   = lane & 15, half = lane >> 4;
#pragma unroll
    for (int r = 0; r < 8; ++r)
        atomicAdd(&outp[(size_t)(u0 + r + 8 * half) * ldo + cc0 + n], acc[r]);
}

// ---- kernel 2: dt = softplus(T@W2 + b); h_new = abar*h + dt*B*x ----------
// 5120 tiles (16 u-tiles x 320 d-tiles), grid 640 blocks * 8 waves. The
// same wave that owns the 16x16 dt tile performs the [16u x 16d x 16n]
// streaming update, so dt never touches HBM.

__global__ void k_gemm2_fused(const float* __restrict__ T,    // [U,R] (ws)
                              const float* __restrict__ W2,   // [R,D]
                              const float* __restrict__ bias, // [D]
                              const float* __restrict__ Bm,   // [U,N] (ws)
                              const float* __restrict__ x,    // [U,D]
                              const float* __restrict__ abar, // [U,D,N]
                              const float* __restrict__ h,    // [U,D,N]
                              float* __restrict__ out) {      // [U,D,N]
    const int lane = threadIdx.x & 31;
    const int wid  = threadIdx.x >> 5;
    const int tile = blockIdx.x * 8 + wid;      // wave-uniform
    const int u0 = (tile & 15) * 16;            // 16 u-tiles
    const int d0 = (tile >> 4) * 16;            // 320 d-tiles

    v8f acc = {};
#pragma unroll
    for (int k = 0; k < R_DIM; k += 32) {       // 5 iters -> 15 WMMA ops
        FragPair a = load_a_frag(T, R_DIM, u0, k, lane);
        FragPair b = load_b_frag(W2, D_DIM, k, d0, lane);
        acc = wmma3(a, b, acc);
    }

    const int n = lane & 15, half = lane >> 4;
    const int d = d0 + n;
    const float bv = bias[d];

#pragma unroll
    for (int r = 0; r < 8; ++r) {
        const int u = u0 + r + 8 * half;
        const float v  = acc[r] + bv;
        const float dt = (v > 20.0f) ? v : log1pf(__expf(v)); // softplus
        const float xv = x[(size_t)u * D_DIM + d];
        const float s  = dt * xv;                              // dt[u,d]*x[u,d]

        const f4* Bv4 = reinterpret_cast<const f4*>(Bm + (size_t)u * N_DIM);
        const size_t base = ((size_t)u * D_DIM + d) * N_DIM;   // 64B aligned
        const f4* ab4 = reinterpret_cast<const f4*>(abar + base);
        const f4* h4  = reinterpret_cast<const f4*>(h + base);
        f4*       o4  = reinterpret_cast<f4*>(out + base);
#pragma unroll
        for (int g = 0; g < 4; ++g) {
            // 252 MB of pure streaming: keep it non-temporal.
            f4 a_ = __builtin_nontemporal_load(ab4 + g);
            f4 h_ = __builtin_nontemporal_load(h4 + g);
            f4 b_ = Bv4[g];                     // hot (16 KB total), cached
            f4 o;
#pragma unroll
            for (int e = 0; e < 4; ++e)
                o[e] = a_[e] * h_[e] + s * b_[e];
            __builtin_nontemporal_store(o, o4 + g);
        }
    }
}

// ---- launch --------------------------------------------------------------

extern "C" void kernel_launch(void* const* d_in, const int* in_sizes, int n_in,
                              void* d_out, int out_size, void* d_ws, size_t ws_size,
                              hipStream_t stream) {
    const float* x    = (const float*)d_in[0];
    const float* W1   = (const float*)d_in[1];  // delta_t_proj_w [D,R]
    const float* W2   = (const float*)d_in[2];  // dt_proj_w      [R,D]
    const float* bias = (const float*)d_in[3];  // dt_proj_b      [D]
    const float* Wb   = (const float*)d_in[4];  // B_proj_w       [D,N]
    const float* abar = (const float*)d_in[5];  // [U,D,N]
    const float* h    = (const float*)d_in[6];  // [U,D,N]
    float* out = (float*)d_out;

    float* T  = (float*)d_ws;                   // [U,R]
    float* Bm = T + U_DIM * R_DIM;              // [U,N]
    const int nz = U_DIM * R_DIM + U_DIM * N_DIM;

    k_zero<<<(nz + 255) / 256, 256, 0, stream>>>(T, nz);
    k_gemm1<<<176, 256, 0, stream>>>(x, W1, Wb, T, Bm);
    k_gemm2_fused<<<640, 256, 0, stream>>>(T, W2, bias, Bm, x, abar, h, out);
}
